// Listener_79439715107129
// MI455X (gfx1250) — matrix operand order repaired and checked by reading, not compiled, not run on hardware
//
#include <hip/hip_runtime.h>
#include <hip/hip_bf16.h>

// ---------------- problem constants ----------------
#define NV   32001
#define NE   512
#define NH   512
#define NLE  512
#define NB   128
#define NT   512
#define NG   2560        // 5*NH

typedef __bf16 bf16_t;
typedef __attribute__((ext_vector_type(16))) __bf16 v16bf;
typedef __attribute__((ext_vector_type(8)))  __bf16 v8bf;
typedef __attribute__((ext_vector_type(8)))  float  v8f;
typedef __attribute__((ext_vector_type(4)))  unsigned u32x4;
typedef __attribute__((ext_vector_type(8)))  unsigned u32x8;

// ---------------- WMMA fragment helpers ----------------
// CDNA5 v_wmma_f32_16x16x32_bf16:  D(16x16,f32) = A(16x32,bf16) x B(32x16,bf16) + C
// A layout: lane = M%16 row; elements 0..7 <-> K = (hi?8:0)+i ; 8..15 <-> K = 16+(hi?8:0)+i
// B layout: lane = N%16 col; elements 0..7 <-> K = (hi?8:0)+i ; 8..15 <-> K = 16+(hi?8:0)+i
// C/D layout: vgpr r <-> M = r + (hi?8:0), N = lane%16

__device__ __forceinline__ v16bf frag_cat(v8bf lo, v8bf hi8) {
  v16bf f;
#pragma unroll
  for (int i = 0; i < 8; ++i) { f[i] = lo[i]; f[8 + i] = hi8[i]; }
  return f;
}

// A fragment from bf16 row-major [rows x lda]
__device__ __forceinline__ v16bf load_a_bf(const bf16_t* A, int lda, int row0, int k0) {
  int lane = threadIdx.x & 31;
  int hi = lane >> 4;
  const bf16_t* p = A + (row0 + (lane & 15)) * lda + k0 + hi * 8;
  v8bf a0 = *(const v8bf*)(p);
  v8bf a1 = *(const v8bf*)(p + 16);
  return frag_cat(a0, a1);
}

// A fragment from an f32 row pointer with in-register f32->bf16 (RNE)
__device__ __forceinline__ v16bf load_a_f32row(const float* row, int k0, int hi) {
  const float* p = row + k0 + hi * 8;
  v16bf f;
#pragma unroll
  for (int i = 0; i < 8; ++i) { f[i] = (bf16_t)p[i]; f[8 + i] = (bf16_t)p[16 + i]; }
  return f;
}

// B fragment from N-major bf16 [n rows x ldb K-cols] (i.e. W[g][k], k contiguous)
__device__ __forceinline__ v16bf load_b(const bf16_t* B, int ldb, int n0, int k0) {
  int lane = threadIdx.x & 31;
  const bf16_t* p = B + (n0 + (lane & 15)) * ldb + k0 + (lane >> 4) * 8;
  v8bf b0 = *(const v8bf*)(p);
  v8bf b1 = *(const v8bf*)(p + 16);
  return frag_cat(b0, b1);
}

__device__ __forceinline__ v8f wmma_bf16(v16bf a, v16bf b, v8f c) {
  return __builtin_amdgcn_wmma_f32_16x16x32_bf16(false, a, false, b, (short)0, c,
                                                 false, false);
}

// ---------------- workspace layout (bytes) ----------------
#define PRE_OFF  ((size_t)0)
#define PRE_SZ   ((size_t)NT * NB * NG * 4)           // 671,088,640
#define I2H_OFF  (PRE_OFF + PRE_SZ)
#define H2H_OFF  (I2H_OFF + (size_t)NG * NE * 2)
#define L0_OFF   (H2H_OFF + (size_t)NG * NH * 2)
#define L1_OFF   (L0_OFF + (size_t)NLE * NH * 2)
#define HBF_OFF  (L1_OFF + (size_t)NLE * NLE * 2)     // 2 x (B x H) bf16 double buffer
#define XSEL_OFF (HBF_OFF + (size_t)2 * NB * NH * 2)
#define Y0_OFF   (XSEL_OFF + (size_t)NB * NH * 4)
#define BAR_OFF  (Y0_OFF + (size_t)NB * NLE * 4)

// ---------------- kernel 0: weight convert + state init ----------------
__global__ __launch_bounds__(256) void k_prep(const float* i2h_w, const float* h2h_w,
                                              const float* lin0_w, const float* lin1_w,
                                              bf16_t* i2h_bf, bf16_t* h2h_bf,
                                              bf16_t* lin0_bf, bf16_t* lin1_bf,
                                              bf16_t* hbuf, unsigned* bar) {
  int i = blockIdx.x * 256 + threadIdx.x;
  if (i < NG * NE) { i2h_bf[i] = (bf16_t)i2h_w[i]; h2h_bf[i] = (bf16_t)h2h_w[i]; }
  if (i < NLE * NH) { lin0_bf[i] = (bf16_t)lin0_w[i]; lin1_bf[i] = (bf16_t)lin1_w[i]; }
  if (i < 2 * NB * NH) hbuf[i] = (bf16_t)0.0f;
  if (i < 2) bar[i] = 0u;
}

// ---------------- kernel 1: pre[t][b][g] = embed[label[b][t]] @ i2h^T + (i2h_b+h2h_b)
// grid (40, 512): x = 64-wide N group, y = t ; 256 threads = 8 waves
// i2h B-chunk staged into LDS with GLOBAL_LOAD_ASYNC_TO_LDS_B128 (ASYNCcnt path)
__global__ __launch_bounds__(256) void k_pre_gemm(const int* label, const float* embed_w,
                                                  const bf16_t* i2h_bf, const float* i2h_b,
                                                  const float* h2h_b, float* pre) {
  __shared__ bf16_t Bs[64 * 136];   // 64 g-rows x 128 K chunk, stride 136 (bank-pad)
  const int t = blockIdx.y;
  const int g0 = blockIdx.x * 64;
  const int tid = threadIdx.x;
  const int wave = tid >> 5, lane = tid & 31, hi = lane >> 4;
  const int brow = wave * 16 + (lane & 15);
  const float* arow = embed_w + (long)label[brow * NT + t] * NE;
  const unsigned ldsb = (unsigned)(uintptr_t)(&Bs[0]);

  v8f acc[4];
#pragma unroll
  for (int nt = 0; nt < 4; ++nt)
#pragma unroll
    for (int r = 0; r < 8; ++r) acc[nt][r] = 0.0f;

  for (int kchunk = 0; kchunk < NE; kchunk += 128) {
    __syncthreads();   // previous chunk fully consumed before overwrite
#pragma unroll
    for (int i = tid * 8; i < 64 * 128; i += 256 * 8) {
      int r = i >> 7, c = i & 127;
      unsigned dst = ldsb + (unsigned)((r * 136 + c) * 2);
      unsigned long long src =
          (unsigned long long)(uintptr_t)(i2h_bf + (size_t)(g0 + r) * NE + kchunk + c);
      asm volatile("global_load_async_to_lds_b128 %0, %1, off"
                   :: "v"(dst), "v"(src) : "memory");
    }
    asm volatile("s_wait_asynccnt 0x0" ::: "memory");
    __syncthreads();

    v16bf a = load_a_f32row(arow, kchunk, hi);
#pragma unroll
    for (int kk = 0; kk < 128; kk += 32) {
      v16bf b0 = load_b(Bs, 136, 0,  kk);
      v16bf b1 = load_b(Bs, 136, 16, kk);
      v16bf b2 = load_b(Bs, 136, 32, kk);
      v16bf b3 = load_b(Bs, 136, 48, kk);
      v16bf an = a;
      if (kk < 96) an = load_a_f32row(arow, kchunk + kk + 32, hi);
      acc[0] = wmma_bf16(a, b0, acc[0]);
      acc[1] = wmma_bf16(a, b1, acc[1]);
      acc[2] = wmma_bf16(a, b2, acc[2]);
      acc[3] = wmma_bf16(a, b3, acc[3]);
      a = an;
    }
  }
#pragma unroll
  for (int nt = 0; nt < 4; ++nt) {
    int g = g0 + nt * 16 + (lane & 15);
    float bias = i2h_b[g] + h2h_b[g];
#pragma unroll
    for (int r = 0; r < 8; ++r) {
      int b = wave * 16 + r + hi * 8;
      pre[((long)t * NB + b) * NG + g] = acc[nt][r] + bias;
    }
  }
}

// ---------------- kernel 2: persistent LSTM recurrence ----------------
// 32 blocks x 256 threads. Block j owns h columns [16j,16j+16): all 5 gate slices of
// h2h are DMA'd once by the Tensor Data Mover into LDS (83 KB of the 320 KB WGP LDS)
// with hardware padding reproducing the 520-element bank-conflict-free stride.
// h is double-buffered bf16 in global; one device-scope barrier per step.
__global__ __launch_bounds__(256) void k_lstm(const float* pre, const bf16_t* h2h_bf,
                                              bf16_t* hbuf, const int* label_len,
                                              float* xsel, unsigned* bar) {
  extern __shared__ bf16_t Ws[];           // [5*16][520]
  const int tid = threadIdx.x, wave = tid >> 5, lane = tid & 31, hi = lane >> 4;
  const int col0 = blockIdx.x * 16;

  // ---- TDM: one 2D tensor descriptor per gate slice (16 rows x 512 K, 2B elems).
  // pad_enable: +4 DWORDs (16B = 8 elems) after every 256 DWORDs (1024B = 512 elems)
  // -> LDS row stride 520 elements, matching load_b(Ws, 520, ...).
  if (tid < 32) {                          // wave 0 issues; TENSORcnt is per-wave
#pragma unroll 1
    for (int gate = 0; gate < 5; ++gate) {
      unsigned long long ga =
          (unsigned long long)(uintptr_t)(h2h_bf + (size_t)(gate * NH + col0) * NH);
      u32x4 d0;
      d0[0] = 1u;                                            // count=1, user descriptor
      d0[1] = (unsigned)(uintptr_t)(&Ws[0]) + (unsigned)(gate * 16 * 520 * 2);
      d0[2] = (unsigned)(ga & 0xFFFFFFFFu);                  // global_addr[31:0]
      d0[3] = (unsigned)((ga >> 32) & 0x01FFFFFFu) | (2u << 30);  // addr[56:32] | type=2
      u32x8 d1;
      d1[0] = (1u << 16) | (1u << 20) | (7u << 22) | (3u << 25); // 2B, pad 4dw/256dw
      d1[1] = (512u & 0xFFFFu) << 16;                        // tensor_dim0 lo16 (=512)
      d1[2] = ((512u >> 16) & 0xFFFFu) | ((2560u & 0xFFFFu) << 16); // dim0 hi | dim1 lo
      d1[3] = ((2560u >> 16) & 0xFFFFu) | (512u << 16);      // dim1 hi | tile_dim0=512
      d1[4] = 16u;                                           // tile_dim1=16, tile_dim2=0
      d1[5] = 512u;                                          // tensor_dim0_stride lo32
      d1[6] = 0u;                                            // stride hi | dim1_stride lo
      d1[7] = 0u;
      asm volatile("tensor_load_to_lds %0, %1" :: "s"(d0), "s"(d1) : "memory");
    }
    __builtin_amdgcn_s_wait_tensorcnt(0);
  }
  __syncthreads();

  int llen[8];
#pragma unroll
  for (int r = 0; r < 8; ++r) llen[r] = label_len[wave * 16 + r + hi * 8];

  v8f c;
#pragma unroll
  for (int r = 0; r < 8; ++r) c[r] = 0.0f;

  unsigned* cnt = bar;
  unsigned* gen = bar + 1;

  for (int t = 0; t < NT; ++t) {
    if (t > 0) {
      // all blocks must finish writing h(t-1) before anyone reads it
      __builtin_amdgcn_fence(__ATOMIC_RELEASE, "agent");
      __syncthreads();
      if (tid == 0) {
        unsigned prev = atomicAdd(cnt, 1u);
        if (prev == 31u) { atomicExch(cnt, 0u); atomicAdd(gen, 1u); }
        else {
          while ((int)atomicAdd(gen, 0u) < t) __builtin_amdgcn_s_sleep(4);
        }
      }
      __syncthreads();
      __builtin_amdgcn_fence(__ATOMIC_ACQUIRE, "agent");
    }

    // s = pre[t] tile
    v8f s[5];
    const float* pt = pre + (long)t * NB * NG;
#pragma unroll
    for (int g = 0; g < 5; ++g) {
      int colg = g * NH + col0 + (lane & 15);
#pragma unroll
      for (int r = 0; r < 8; ++r) {
        int b = wave * 16 + r + hi * 8;
        s[g][r] = pt[(long)b * NG + colg];
      }
    }
    if (t + 1 < NT)  // warm next step's pre tile toward the caches
      __builtin_prefetch(pt + (long)NB * NG + (long)(wave * 16 + (lane & 15)) * NG + col0, 0, 1);

    // s += h(t-1) @ h2h^T  (A pipelined from global bf16, B from LDS-resident weights)
    const bf16_t* hsrc = hbuf + (size_t)(t & 1) * NB * NH;
    v16bf a = load_a_bf(hsrc, NH, wave * 16, 0);
#pragma unroll 4
    for (int kc = 0; kc < NH; kc += 32) {
      v16bf b0 = load_b(Ws, 520, 0,  kc);
      v16bf b1 = load_b(Ws, 520, 16, kc);
      v16bf b2 = load_b(Ws, 520, 32, kc);
      v16bf b3 = load_b(Ws, 520, 48, kc);
      v16bf b4 = load_b(Ws, 520, 64, kc);
      v16bf an = a;
      if (kc + 32 < NH) an = load_a_bf(hsrc, NH, wave * 16, kc + 32);
      s[0] = wmma_bf16(a, b0, s[0]);
      s[1] = wmma_bf16(a, b1, s[1]);
      s[2] = wmma_bf16(a, b2, s[2]);
      s[3] = wmma_bf16(a, b3, s[3]);
      s[4] = wmma_bf16(a, b4, s[4]);
      a = an;
    }

    // gates + state update (registers), write h(t) to the other buffer
    bf16_t* hdst = hbuf + (size_t)((t + 1) & 1) * NB * NH;
    const int colj = col0 + (lane & 15);
#pragma unroll
    for (int r = 0; r < 8; ++r) {
      float ig = 1.0f / (1.0f + __expf(-s[0][r]));
      float fg = 1.0f / (1.0f + __expf(-s[1][r]));
      float og = 1.0f / (1.0f + __expf(-s[2][r]));
      float in_t = fmaxf(s[3][r], s[4][r]);
      float nc = fg * c[r] + ig * in_t;
      c[r] = nc;
      float nh = og * tanhf(nc);
      int b = wave * 16 + r + hi * 8;
      hdst[b * NH + colj] = (bf16_t)nh;
      if (t == llen[r]) xsel[b * NH + colj] = nh;   // outs[label_len[b], b]
    }
  }
}

// ---------------- kernel 3: linear + batchnorm (+optional relu) ----------------
// 32 blocks: block j owns 16 output cols over the full batch -> BN is block-local.
__global__ __launch_bounds__(256) void k_head(const float* X, const bf16_t* Wbf,
                                              const float* bias, const float* gamma,
                                              const float* beta, float* out, int relu) {
  __shared__ float S[NB * 17];
  __shared__ float Ms[16], Rs[16];
  const int tid = threadIdx.x, wave = tid >> 5, lane = tid & 31, hi = lane >> 4;
  const int col0 = blockIdx.x * 16;
  const int cl = lane & 15;

  v8f acc;
#pragma unroll
  for (int r = 0; r < 8; ++r) acc[r] = 0.0f;

  const float* xrow = X + (wave * 16 + (lane & 15)) * NH;
  v16bf a = load_a_f32row(xrow, 0, hi);
#pragma unroll 4
  for (int kc = 0; kc < NH; kc += 32) {
    v16bf b = load_b(Wbf, NH, col0, kc);
    v16bf an = a;
    if (kc + 32 < NH) an = load_a_f32row(xrow, kc + 32, hi);
    acc = wmma_bf16(a, b, acc);
    a = an;
  }

  float bi = bias[col0 + cl];
#pragma unroll
  for (int r = 0; r < 8; ++r) {
    int b = wave * 16 + r + hi * 8;
    S[b * 17 + cl] = acc[r] + bi;
  }
  __syncthreads();
  if (tid < 16) {
    float sum = 0.0f, sq = 0.0f;
    for (int b = 0; b < NB; ++b) { float v = S[b * 17 + tid]; sum += v; sq += v * v; }
    float m = sum * (1.0f / NB);
    float var = sq * (1.0f / NB) - m * m;
    Ms[tid] = m;
    Rs[tid] = rsqrtf(var + 1e-5f);
  }
  __syncthreads();
  float g = gamma[col0 + cl], be = beta[col0 + cl];
#pragma unroll
  for (int r = 0; r < 8; ++r) {
    int b = wave * 16 + r + hi * 8;
    float y = (S[b * 17 + cl] - Ms[cl]) * Rs[cl] * g + be;
    if (relu) y = fmaxf(y, 0.0f);
    out[b * NLE + col0 + cl] = y;
  }
}

// ---------------- host launcher ----------------
extern "C" void kernel_launch(void* const* d_in, const int* in_sizes, int n_in,
                              void* d_out, int out_size, void* d_ws, size_t ws_size,
                              hipStream_t stream) {
  (void)in_sizes; (void)n_in; (void)out_size; (void)ws_size;
  const int*   label     = (const int*)d_in[0];
  const int*   label_len = (const int*)d_in[1];
  const float* embed_w   = (const float*)d_in[2];
  const float* i2h_w     = (const float*)d_in[3];
  const float* i2h_b     = (const float*)d_in[4];
  const float* h2h_w     = (const float*)d_in[5];
  const float* h2h_b     = (const float*)d_in[6];
  const float* lin0_w    = (const float*)d_in[7];
  const float* lin0_b    = (const float*)d_in[8];
  const float* bn0_g     = (const float*)d_in[9];
  const float* bn0_b     = (const float*)d_in[10];
  const float* lin1_w    = (const float*)d_in[11];
  const float* lin1_b    = (const float*)d_in[12];
  const float* bn1_g     = (const float*)d_in[13];
  const float* bn1_b     = (const float*)d_in[14];

  char* ws = (char*)d_ws;
  float*    pre     = (float*)(ws + PRE_OFF);
  bf16_t*   i2h_bf  = (bf16_t*)(ws + I2H_OFF);
  bf16_t*   h2h_bf  = (bf16_t*)(ws + H2H_OFF);
  bf16_t*   lin0_bf = (bf16_t*)(ws + L0_OFF);
  bf16_t*   lin1_bf = (bf16_t*)(ws + L1_OFF);
  bf16_t*   hbuf    = (bf16_t*)(ws + HBF_OFF);
  float*    xsel    = (float*)(ws + XSEL_OFF);
  float*    y0      = (float*)(ws + Y0_OFF);
  unsigned* bar     = (unsigned*)(ws + BAR_OFF);

  k_prep<<<(NG * NE + 255) / 256, 256, 0, stream>>>(
      i2h_w, h2h_w, lin0_w, lin1_w, i2h_bf, h2h_bf, lin0_bf, lin1_bf, hbuf, bar);

  k_pre_gemm<<<dim3(NG / 64, NT), 256, 0, stream>>>(
      label, embed_w, i2h_bf, i2h_b, h2h_b, pre);

  k_lstm<<<32, 256, 5 * 16 * 520 * (int)sizeof(bf16_t), stream>>>(
      pre, h2h_bf, hbuf, label_len, xsel, bar);

  k_head<<<32, 256, 0, stream>>>(xsel, lin0_bf, lin0_b, bn0_g, bn0_b, y0, 1);
  k_head<<<32, 256, 0, stream>>>(y0, lin1_bf, lin1_b, bn1_g, bn1_b, (float*)d_out, 0);
}
